// MEMC_Net_s_79388175499806
// MI455X (gfx1250) — matrix operand order repaired and checked by reading, not compiled, not run on hardware
//
#include <hip/hip_runtime.h>
#include <hip/hip_bf16.h>

typedef __attribute__((ext_vector_type(16))) _Float16 v16h;
typedef __attribute__((ext_vector_type(8)))  _Float16 v8h;
typedef __attribute__((ext_vector_type(8)))  float    v8f;

#define TPB 256

__device__ __forceinline__ int clampi(int v, int lo, int hi) {
  return v < lo ? lo : (v > hi ? hi : v);
}

// ---------------------------------------------------------------------------
// Implicit-GEMM 3x3 SAME conv via V_WMMA_F32_16X16X32_F16, LDS-staged inputs.
// Activations: NHWC f16, channels padded to Cinp = CHUNKS*32 (pads are zero).
// Weights: fragment-ready f16, afrag[((cot*9+tap)*CHUNKS+ch)*512 + lane*16 + e].
// One wave per 16-pixel row tile x exactly TM 16-cout tiles.
// Per chunk: stage 3x18x32 halo tile into LDS (clamped loads + zero-select),
// then 9 taps read B fragments straight from LDS (aligned 32B ds reads).
// Single-wave workgroup => no barrier; LDS ops are in-order per wave (DScnt).
// ---------------------------------------------------------------------------
template <int CHUNKS, int TM>
__global__ __launch_bounds__(32) void conv3x3_wmma_lds(
    const _Float16* __restrict__ xh, const _Float16* __restrict__ afrag,
    const float* __restrict__ biasp,
    _Float16* __restrict__ yh, float* __restrict__ yf,
    int Bn, int H, int W, int Coutp, int Cout, int relu, int outmode)
{
  __shared__ _Float16 tile[3 * 18 * 32];         // 3456 B
  const int Cinp = CHUNKS * 32;
  const int lane = threadIdx.x;
  const int l15  = lane & 15;
  const int hi   = (lane >> 4) & 1;

  const int tilesPerRow = W >> 4;
  int tIdx = blockIdx.x;
  int pw0 = (tIdx % tilesPerRow) * 16;
  int t2  = tIdx / tilesPerRow;
  int ph  = t2 % H;
  int b   = t2 / H;
  const int tile0 = blockIdx.y * TM;

  const v16h* Afr = (const v16h*)afrag;
  v16h zero = {};
  v8f acc[TM] = {};

  for (int ch = 0; ch < CHUNKS; ++ch) {
    // ---- stage 3 rows x 18 px x 32 ch of this chunk into LDS ----
#pragma unroll
    for (int u0 = 0; u0 < 4; ++u0) {
      int u = u0 * 32 + lane;                     // 108 units of 16 halfs
      if (u < 108) {
        int row  = u / 36;                        // 0..2
        int rem  = u - row * 36;
        int px   = rem >> 1;                      // 0..17
        int half = rem & 1;
        int gy = ph + row - 1;
        int gx = pw0 + px - 1;
        bool ok = (gy >= 0) & (gy < H) & (gx >= 0) & (gx < W);
        int gyc = clampi(gy, 0, H - 1);
        int gxc = clampi(gx, 0, W - 1);
        v16h v = *(const v16h*)(xh + ((long)(b * H + gyc) * W + gxc) * Cinp
                                + ch * 32 + half * 16);
        if (!ok) v = zero;
        *(v16h*)(&tile[(row * 18 + px) * 32 + half * 16]) = v;
      }
    }
    __builtin_prefetch(&Afr[(((long)tile0 * 9) * CHUNKS + ch + 1) * 32 + lane], 0, 1);

    // ---- 9 taps from LDS ----
    for (int tap = 0; tap < 9; ++tap) {
      int dh = tap / 3;
      int dw = tap - dh * 3;                      // 0..2
      v16h bv = *(const v16h*)(&tile[((dh * 18) + l15 + dw) * 32 + hi * 16]);
#pragma unroll
      for (int t = 0; t < TM; ++t) {
        v16h av = Afr[(((long)(tile0 + t) * 9 + tap) * CHUNKS + ch) * 32 + lane];
        acc[t] = __builtin_amdgcn_wmma_f32_16x16x32_f16(
            false, av, false, bv, (short)0, acc[t], false, false);
      }
    }
  }

  int pw = pw0 + l15;
  if (outmode == 0) {
    long pixo = ((long)(b * H + ph) * W + pw) * Coutp;
#pragma unroll
    for (int t = 0; t < TM; ++t) {
      int co0 = (tile0 + t) * 16;
      v8h ov;
#pragma unroll
      for (int r = 0; r < 8; ++r) {
        float val = acc[t][r] + biasp[co0 + hi * 8 + r];
        if (relu) val = fmaxf(val, 0.f);
        ov[r] = (_Float16)val;
      }
      *(v8h*)(yh + pixo + co0 + hi * 8) = ov;
    }
  } else {
    int HW = H * W;
#pragma unroll
    for (int t = 0; t < TM; ++t) {
      int co0 = (tile0 + t) * 16;
#pragma unroll
      for (int r = 0; r < 8; ++r) {
        int mo = co0 + hi * 8 + r;
        if (mo < Cout) {
          float val = acc[t][r] + biasp[mo];
          if (relu) val = fmaxf(val, 0.f);
          yf[((long)b * Cout + mo) * HW + (long)ph * W + pw] = val;
        }
      }
    }
  }
}

// Direct-global fallback (used only when W%16 != 0, i.e. the 8x8 level).
template <int CHUNKS, int TM>
__global__ __launch_bounds__(32) void conv3x3_wmma_dir(
    const _Float16* __restrict__ xh, const _Float16* __restrict__ afrag,
    const float* __restrict__ biasp,
    _Float16* __restrict__ yh, float* __restrict__ yf,
    int Bn, int H, int W, int Coutp, int Cout, int relu, int outmode)
{
  const int Cinp = CHUNKS * 32;
  const int lane = threadIdx.x;
  const int l15  = lane & 15;
  const int hi   = (lane >> 4) & 1;
  const int HW   = H * W;

  const long n0    = (long)blockIdx.x * 16;
  const int  tile0 = blockIdx.y * TM;

  long p   = n0 + l15;
  int  b   = (int)(p / HW);
  int  rem = (int)(p - (long)b * HW);
  int  ph  = rem / W;
  int  pw  = rem - ph * W;

  const v16h* Afr = (const v16h*)afrag;
  v16h bzero = {};
  v8f acc[TM] = {};

  for (int tap = 0; tap < 9; ++tap) {
    int dh = tap / 3 - 1;
    int dw = tap - (tap / 3) * 3 - 1;
    int hh = ph + dh, ww = pw + dw;
    bool inb = (hh >= 0) & (hh < H) & (ww >= 0) & (ww < W);
    int hhc = clampi(hh, 0, H - 1);
    int wwc = clampi(ww, 0, W - 1);
    const _Float16* xb = xh + ((long)(b * H + hhc) * W + wwc) * Cinp + hi * 16;

#pragma unroll 2
    for (int ch = 0; ch < CHUNKS; ++ch) {
      v16h bv = *(const v16h*)(xb + ch * 32);
      if (!inb) bv = bzero;
#pragma unroll
      for (int t = 0; t < TM; ++t) {
        v16h av = Afr[(((long)(tile0 + t) * 9 + tap) * CHUNKS + ch) * 32 + lane];
        acc[t] = __builtin_amdgcn_wmma_f32_16x16x32_f16(
            false, av, false, bv, (short)0, acc[t], false, false);
      }
    }
  }

  if (outmode == 0) {
    long pixo = ((long)(b * H + ph) * W + pw) * Coutp;
#pragma unroll
    for (int t = 0; t < TM; ++t) {
      int co0 = (tile0 + t) * 16;
      v8h ov;
#pragma unroll
      for (int r = 0; r < 8; ++r) {
        float val = acc[t][r] + biasp[co0 + hi * 8 + r];
        if (relu) val = fmaxf(val, 0.f);
        ov[r] = (_Float16)val;
      }
      *(v8h*)(yh + pixo + co0 + hi * 8) = ov;
    }
  } else {
#pragma unroll
    for (int t = 0; t < TM; ++t) {
      int co0 = (tile0 + t) * 16;
#pragma unroll
      for (int r = 0; r < 8; ++r) {
        int mo = co0 + hi * 8 + r;
        if (mo < Cout) {
          float val = acc[t][r] + biasp[mo];
          if (relu) val = fmaxf(val, 0.f);
          yf[((long)b * Cout + mo) * HW + (long)ph * W + pw] = val;
        }
      }
    }
  }
}

template <int C>
static void launch_conv(hipStream_t s, int TM, bool useLds, dim3 g,
                        const _Float16* x, const _Float16* af, const float* bp,
                        _Float16* yh, float* yf,
                        int B, int H, int W, int Coutp, int Cout, int relu, int mode)
{
  if (useLds) {
    if (TM == 1)      conv3x3_wmma_lds<C, 1><<<g, 32, 0, s>>>(x, af, bp, yh, yf, B, H, W, Coutp, Cout, relu, mode);
    else if (TM == 2) conv3x3_wmma_lds<C, 2><<<g, 32, 0, s>>>(x, af, bp, yh, yf, B, H, W, Coutp, Cout, relu, mode);
    else              conv3x3_wmma_lds<C, 4><<<g, 32, 0, s>>>(x, af, bp, yh, yf, B, H, W, Coutp, Cout, relu, mode);
  } else {
    if (TM == 1)      conv3x3_wmma_dir<C, 1><<<g, 32, 0, s>>>(x, af, bp, yh, yf, B, H, W, Coutp, Cout, relu, mode);
    else if (TM == 2) conv3x3_wmma_dir<C, 2><<<g, 32, 0, s>>>(x, af, bp, yh, yf, B, H, W, Coutp, Cout, relu, mode);
    else              conv3x3_wmma_dir<C, 4><<<g, 32, 0, s>>>(x, af, bp, yh, yf, B, H, W, Coutp, Cout, relu, mode);
  }
}

// OIHW f32 -> fragment-ready f16 (zero-padded in K and M)
__global__ void prepack_afrag_k(const float* __restrict__ w, _Float16* __restrict__ afrag,
                                int Cin, int Cout, int nchunk, int nCot)
{
  long total = (long)nCot * 9 * nchunk * 512;
  long idx = blockIdx.x * (long)blockDim.x + threadIdx.x;
  if (idx >= total) return;
  int e    = (int)(idx & 15);
  int lane = (int)((idx >> 4) & 31);
  long f   = idx >> 9;
  int ch   = (int)(f % nchunk);
  long t2  = f / nchunk;
  int tap  = (int)(t2 % 9);
  int cot  = (int)(t2 / 9);
  int v = e >> 1, hp = e & 1, hi = lane >> 4, l15 = lane & 15;
  int kA = (hi ? 16 : 0) + (v << 1) + hp;
  int ci = ch * 32 + kA;
  int m  = cot * 16 + l15;
  _Float16 val = (_Float16)0.f;
  if (ci < Cin && m < Cout) val = (_Float16)w[((long)m * Cin + ci) * 9 + tap];
  afrag[idx] = val;
}

__global__ void prepack_bias_k(const float* __restrict__ b, float* __restrict__ bp,
                               int Cout, int Coutp)
{
  int i = blockIdx.x * blockDim.x + threadIdx.x;
  if (i < Coutp) bp[i] = (i < Cout) ? b[i] : 0.f;
}

// frames (2,B,3,H,W) f32 -> x (B,H,W,32) f16, channels 6..31 zero
__global__ void pack_frames_f16_k(const float* __restrict__ frames, _Float16* __restrict__ xh,
                                  int B, int H, int W)
{
  const int Cp = 32;
  long HW = (long)H * W;
  long total = (long)B * HW * Cp;
  long i = blockIdx.x * (long)blockDim.x + threadIdx.x;
  if (i >= total) return;
  int c  = (int)(i % Cp);
  long t = i / Cp;
  long hw = t % HW;
  int b  = (int)(t / HW);
  float v = 0.f;
  if (c < 6) {
    int fidx = c / 3, cc = c - fidx * 3;
    v = frames[(((long)fidx * B + b) * 3 + cc) * HW + hw];
  }
  xh[i] = (_Float16)v;
}

__global__ void maxpool2_f16_k(const _Float16* __restrict__ x, _Float16* __restrict__ y,
                               int B, int Cp, int H, int W)
{
  int oH = H >> 1, oW = W >> 1;
  long total = (long)B * oH * oW * Cp;
  long i = blockIdx.x * (long)blockDim.x + threadIdx.x;
  if (i >= total) return;
  int c  = (int)(i % Cp);
  long t = i / Cp;
  int ox = (int)(t % oW); t /= oW;
  int oy = (int)(t % oH);
  int b  = (int)(t / oH);
  const _Float16* xp = x + (((long)b * H + oy * 2) * W + ox * 2) * Cp + c;
  float v0 = (float)xp[0], v1 = (float)xp[Cp];
  float v2 = (float)xp[(long)W * Cp], v3 = (float)xp[(long)W * Cp + Cp];
  y[i] = (_Float16)fmaxf(fmaxf(v0, v1), fmaxf(v2, v3));
}

// y = bilinear_up2(x) + skip (NHWC f16, half-pixel centers, edge clamp)
__global__ void up2_add_f16_k(const _Float16* __restrict__ x, const _Float16* __restrict__ skip,
                              _Float16* __restrict__ y, int B, int Cp, int H, int W)
{
  int oH = H * 2, oW = W * 2;
  long total = (long)B * oH * oW * Cp;
  long i = blockIdx.x * (long)blockDim.x + threadIdx.x;
  if (i >= total) return;
  int c  = (int)(i % Cp);
  long t = i / Cp;
  int ox = (int)(t % oW); t /= oW;
  int oy = (int)(t % oH);
  int b  = (int)(t / oH);
  float sy = (oy + 0.5f) * 0.5f - 0.5f;
  float sx = (ox + 0.5f) * 0.5f - 0.5f;
  int y0 = (int)floorf(sy); float fy = sy - y0;
  int x0 = (int)floorf(sx); float fx = sx - x0;
  int y0c = clampi(y0, 0, H - 1), y1c = clampi(y0 + 1, 0, H - 1);
  int x0c = clampi(x0, 0, W - 1), x1c = clampi(x0 + 1, 0, W - 1);
  const _Float16* xp = x + (long)b * H * W * Cp + c;
  float v00 = (float)xp[((long)y0c * W + x0c) * Cp];
  float v01 = (float)xp[((long)y0c * W + x1c) * Cp];
  float v10 = (float)xp[((long)y1c * W + x0c) * Cp];
  float v11 = (float)xp[((long)y1c * W + x1c) * Cp];
  float v = (1.f - fy) * ((1.f - fx) * v00 + fx * v01)
          +        fy  * ((1.f - fx) * v10 + fx * v11);
  y[i] = (_Float16)(v + (float)skip[i]);
}

__global__ void fill_k(float* __restrict__ p, long n, float v)
{
  long i = blockIdx.x * (long)blockDim.x + threadIdx.x;
  if (i < n) p[i] = v;
}

__global__ void flow_scatter_k(const float* __restrict__ flow, float scale,
                               float* sx, float* sy, float* cnt, int B, int H, int W)
{
  int HW = H * W;
  long total = (long)B * HW;
  long i = blockIdx.x * (long)blockDim.x + threadIdx.x;
  if (i >= total) return;
  int hw = (int)(i % HW);
  int b  = (int)(i / HW);
  int h = hw / W, w = hw - (hw / W) * W;
  float fx = flow[((long)b * 2 + 0) * HW + hw] * scale;
  float fy = flow[((long)b * 2 + 1) * HW + hw] * scale;
  float x2 = w + fx, y2 = h + fy;
  float valid = (x2 >= 0.f && x2 <= (float)(W - 1) &&
                 y2 >= 0.f && y2 <= (float)(H - 1)) ? 1.f : 0.f;
  int ix = (int)floorf(x2), iy = (int)floorf(y2);
#pragma unroll
  for (int dy = 0; dy < 2; ++dy)
#pragma unroll
    for (int dx = 0; dx < 2; ++dx) {
      int jx = clampi(ix + dx, 0, W - 1);
      int jy = clampi(iy + dy, 0, H - 1);
      long o = (long)b * HW + (long)jy * W + jx;
      atomicAdd(&sx[o], -fx * valid);
      atomicAdd(&sy[o], -fy * valid);
      atomicAdd(&cnt[o], valid);
    }
}

__global__ void flow_norm_k(const float* __restrict__ sx, const float* __restrict__ sy,
                            const float* __restrict__ cnt, float* __restrict__ out,
                            int B, int HW)
{
  long total = (long)B * HW;
  long i = blockIdx.x * (long)blockDim.x + threadIdx.x;
  if (i >= total) return;
  int hw = (int)(i % HW);
  int b  = (int)(i / HW);
  float c = cnt[i];
  float s = fmaxf(c, 1.f);
  out[((long)b * 2 + 0) * HW + hw] = (c > 0.f) ? sx[i] / s : 0.f;
  out[((long)b * 2 + 1) * HW + hw] = (c > 0.f) ? sy[i] / s : 0.f;
}

__global__ void filt_interp_k(const float* __restrict__ img, const float* __restrict__ flow,
                              const float* __restrict__ filt, float* __restrict__ out,
                              int B, int H, int W)
{
  int HW = H * W;
  long total = (long)B * HW;
  long i = blockIdx.x * (long)blockDim.x + threadIdx.x;
  if (i >= total) return;
  int hw = (int)(i % HW);
  int b  = (int)(i / HW);
  int h = hw / W, w = hw - (hw / W) * W;
  float fx = flow[((long)b * 2 + 0) * HW + hw];
  float fy = flow[((long)b * 2 + 1) * HW + hw];
  float x2 = w + fx, y2 = h + fy;
  int ix = (int)floorf(x2), iy = (int)floorf(y2);
  float a  = x2 - ix, bb = y2 - iy;
  int ixL = ix - 1, iyT = iy - 1;
  const float* ib = img + (long)b * 3 * HW;
  float o[3] = {0.f, 0.f, 0.f};
  float w00 = (1.f - a) * (1.f - bb), w10 = a * (1.f - bb);
  float w01 = (1.f - a) * bb,         w11 = a * bb;
  for (int fj = 0; fj < 4; ++fj)
    for (int fi = 0; fi < 4; ++fi) {
      int gx = ixL + fi, gy = iyT + fj;
      int x0 = clampi(gx, 0, W - 1), x1 = clampi(gx + 1, 0, W - 1);
      int y0 = clampi(gy, 0, H - 1), y1 = clampi(gy + 1, 0, H - 1);
      float f = filt[((long)b * 16 + fj * 4 + fi) * HW + hw];
#pragma unroll
      for (int c = 0; c < 3; ++c) {
        const float* ic = ib + (long)c * HW;
        float val = ic[y0 * W + x0] * w00 + ic[y0 * W + x1] * w10
                  + ic[y1 * W + x0] * w01 + ic[y1 * W + x1] * w11;
        o[c] += f * val;
      }
    }
#pragma unroll
  for (int c = 0; c < 3; ++c)
    out[((long)b * 3 + c) * HW + hw] = o[c];
}

__global__ void avg2_k(const float* __restrict__ a, const float* __restrict__ b,
                       float* __restrict__ y, long n)
{
  long i = blockIdx.x * (long)blockDim.x + threadIdx.x;
  if (i < n) y[i] = 0.5f * a[i] + 0.5f * b[i];
}

__global__ void add_k(const float* __restrict__ a, const float* __restrict__ b,
                      float* __restrict__ y, long n)
{
  long i = blockIdx.x * (long)blockDim.x + threadIdx.x;
  if (i < n) y[i] = a[i] + b[i];
}

// f32 NCHW src -> f16 NHWC dst (stride Cp) at channel offset cd0
__global__ void copy_ch_nhwc_k(const float* __restrict__ src, int cs0, int Csrc,
                               _Float16* __restrict__ dst, int cd0, int Cp,
                               int nC, int B, int HW)
{
  long total = (long)B * nC * HW;
  long i = blockIdx.x * (long)blockDim.x + threadIdx.x;
  if (i >= total) return;
  int hw = (int)(i % HW);
  long t = i / HW;
  int c = (int)(t % nC);
  int b = (int)(t / nC);
  dst[((long)b * HW + hw) * Cp + cd0 + c] =
      (_Float16)src[((long)b * Csrc + cs0 + c) * HW + hw];
}

// ---------------------------------------------------------------------------

static inline int ceil16(int c) { return (c + 15) & ~15; }
static inline int ceil32(int c) { return (c + 31) & ~31; }

extern "C" void kernel_launch(void* const* d_in, const int* in_sizes, int n_in,
                              void* d_out, int out_size, void* d_ws, size_t ws_size,
                              hipStream_t stream)
{
  const int B = 4, H = 256, W = 256, HW = H * W;

  const float* frames   = (const float*)d_in[0];
  const float* flow_fwd = (const float*)d_in[1];
  const float* flow_bwd = (const float*)d_in[2];

  static const int monoC[12][2] = {{6,16},{16,32},{32,64},{64,128},{128,256},{256,512},
                                   {512,512},{512,256},{256,128},{128,64},{64,32},{32,16}};
  static const int rectC[8][2]  = {{41,64},{64,64},{64,64},{64,64},{64,64},{64,64},{64,64},{64,3}};

  const float *mw[12], *mb[12], *bw[4], *bb4[4], *rw[8], *rb[8];
  {
    int idx = 3;
    bool alpha = (n_in > 3 && in_sizes[3] == 16 * 16 * 9);  // br1 first (alphabetical)
    if (alpha) {
      for (int i = 0; i < 4; ++i) { bw[i] = (const float*)d_in[idx++]; bb4[i] = (const float*)d_in[idx++]; }
      for (int i = 0; i < 12; ++i){ mw[i] = (const float*)d_in[idx++]; mb[i]  = (const float*)d_in[idx++]; }
      for (int i = 0; i < 8; ++i) { rw[i] = (const float*)d_in[idx++]; rb[i]  = (const float*)d_in[idx++]; }
    } else {
      for (int i = 0; i < 12; ++i){ mw[i] = (const float*)d_in[idx++]; mb[i]  = (const float*)d_in[idx++]; }
      for (int i = 0; i < 4; ++i) { bw[i] = (const float*)d_in[idx++]; bb4[i] = (const float*)d_in[idx++]; }
      for (int i = 0; i < 8; ++i) { rw[i] = (const float*)d_in[idx++]; rb[i]  = (const float*)d_in[idx++]; }
    }
  }

  float* base = (float*)d_ws;
  size_t ofs = 0;
  auto allocf = [&](size_t n) -> float* {
    float* p = base + ofs;
    ofs += (n + 63) & ~(size_t)63;
    return p;
  };
  auto alloch = [&](size_t nhalf) -> _Float16* {
    return (_Float16*)allocf((nhalf + 1) / 2);
  };

  _Float16* x6h   = alloch((size_t)B * HW * 32);
  _Float16* pingh = alloch((size_t)B * HW * 64);
  _Float16* pongh = alloch((size_t)B * HW * 64);
  _Float16* s1h = alloch((size_t)B * 256 * 256 * 32);
  _Float16* s2h = alloch((size_t)B * 128 * 128 * 64);
  _Float16* s3h = alloch((size_t)B * 64  * 64  * 128);
  _Float16* s4h = alloch((size_t)B * 32  * 32  * 256);
  _Float16* s5h = alloch((size_t)B * 16  * 16  * 512);
  _Float16* feath = alloch((size_t)B * HW * 32);
  _Float16* rinh  = alloch((size_t)B * HW * 64);
  float* filt0 = allocf((size_t)B * 16 * HW);
  float* filt1 = allocf((size_t)B * 16 * HW);
  float* off0  = allocf((size_t)B * 2 * HW);
  float* off1  = allocf((size_t)B * 2 * HW);
  float* sx    = allocf((size_t)B * HW);
  float* sy    = allocf((size_t)B * HW);
  float* cnt   = allocf((size_t)B * HW);
  float* w0b   = allocf((size_t)B * 3 * HW);
  float* w2b   = allocf((size_t)B * 3 * HW);
  float* r3    = allocf((size_t)B * 3 * HW);

  int mCinp[12], mCoutp[12], bCinp[4], bCoutp[4], rCinp[8], rCoutp[8];
  for (int i = 0; i < 12; ++i) { mCinp[i] = ceil32(monoC[i][0]); mCoutp[i] = ceil32(monoC[i][1]); }
  for (int i = 0; i < 4;  ++i) { bCinp[i] = 32; bCoutp[i] = (i & 1) ? ceil16(16) : ceil32(16); }
  for (int i = 0; i < 8;  ++i) { rCinp[i] = ceil32(rectC[i][0]); rCoutp[i] = (i == 7) ? ceil16(3) : ceil32(rectC[i][1]); }

  _Float16 *mpk[12], *bpk[4], *rpk[8];
  float *mbp[12], *bbp[4], *rbp[8];
  for (int i = 0; i < 12; ++i) { mpk[i] = alloch((size_t)mCoutp[i] * mCinp[i] * 9); mbp[i] = allocf(mCoutp[i]); }
  for (int i = 0; i < 4;  ++i) { bpk[i] = alloch((size_t)bCoutp[i] * bCinp[i] * 9); bbp[i] = allocf(bCoutp[i]); }
  for (int i = 0; i < 8;  ++i) { rpk[i] = alloch((size_t)rCoutp[i] * rCinp[i] * 9); rbp[i] = allocf(rCoutp[i]); }

  auto eb = [](long n) -> unsigned { return (unsigned)((n + TPB - 1) / TPB); };

  auto prepack = [&](const float* w, const float* b, _Float16* af, float* bp,
                     int Cin, int Cout, int Cinp, int Coutp) {
    int nchunk = Cinp / 32, nCot = Coutp / 16;
    long n = (long)nCot * 9 * nchunk * 512;
    prepack_afrag_k<<<eb(n), TPB, 0, stream>>>(w, af, Cin, Cout, nchunk, nCot);
    prepack_bias_k<<<eb(Coutp), TPB, 0, stream>>>(b, bp, Cout, Coutp);
  };
  for (int i = 0; i < 12; ++i) prepack(mw[i], mb[i], mpk[i], mbp[i], monoC[i][0], monoC[i][1], mCinp[i], mCoutp[i]);
  for (int i = 0; i < 4;  ++i) prepack(bw[i], bb4[i], bpk[i], bbp[i], 16, 16, bCinp[i], bCoutp[i]);
  for (int i = 0; i < 8;  ++i) prepack(rw[i], rb[i], rpk[i], rbp[i], rectC[i][0], rectC[i][1], rCinp[i], rCoutp[i]);

  auto convL = [&](const _Float16* xin, const _Float16* af, const float* bp,
                   _Float16* yh, float* yf, int h, int w,
                   int Cinp, int Coutp, int Cout, int relu, int mode) {
    long N = (long)B * h * w;
    int nTiles = Coutp / 16;
    int TM = (nTiles % 4 == 0) ? 4 : ((nTiles % 2 == 0) ? 2 : 1);
    bool useLds = (w % 16 == 0);
    dim3 g((unsigned)(N / 16), (unsigned)(nTiles / TM));
    switch (Cinp / 32) {
      case 1:  launch_conv<1> (stream, TM, useLds, g, xin, af, bp, yh, yf, B, h, w, Coutp, Cout, relu, mode); break;
      case 2:  launch_conv<2> (stream, TM, useLds, g, xin, af, bp, yh, yf, B, h, w, Coutp, Cout, relu, mode); break;
      case 4:  launch_conv<4> (stream, TM, useLds, g, xin, af, bp, yh, yf, B, h, w, Coutp, Cout, relu, mode); break;
      case 8:  launch_conv<8> (stream, TM, useLds, g, xin, af, bp, yh, yf, B, h, w, Coutp, Cout, relu, mode); break;
      default: launch_conv<16>(stream, TM, useLds, g, xin, af, bp, yh, yf, B, h, w, Coutp, Cout, relu, mode); break;
    }
  };
  auto pool = [&](const _Float16* x, _Float16* y, int Cp, int h, int w) {
    long n = (long)B * Cp * (h / 2) * (w / 2);
    maxpool2_f16_k<<<eb(n), TPB, 0, stream>>>(x, y, B, Cp, h, w);
  };
  auto upadd = [&](const _Float16* x, const _Float16* skip, _Float16* y, int Cp, int h, int w) {
    long n = (long)B * Cp * (2 * h) * (2 * w);
    up2_add_f16_k<<<eb(n), TPB, 0, stream>>>(x, skip, y, B, Cp, h, w);
  };

  // ---- mono U-Net ----
  pack_frames_f16_k<<<eb((long)B * HW * 32), TPB, 0, stream>>>(frames, x6h, B, H, W);
  convL(x6h,   mpk[0], mbp[0], pingh, nullptr, 256, 256, mCinp[0], mCoutp[0], 16, 1, 0);
  convL(pingh, mpk[1], mbp[1], s1h,   nullptr, 256, 256, mCinp[1], mCoutp[1], 32, 1, 0);
  pool(s1h, pingh, 32, 256, 256);
  convL(pingh, mpk[2], mbp[2], s2h,   nullptr, 128, 128, mCinp[2], mCoutp[2], 64, 1, 0);
  pool(s2h, pingh, 64, 128, 128);
  convL(pingh, mpk[3], mbp[3], s3h,   nullptr, 64, 64, mCinp[3], mCoutp[3], 128, 1, 0);
  pool(s3h, pingh, 128, 64, 64);
  convL(pingh, mpk[4], mbp[4], s4h,   nullptr, 32, 32, mCinp[4], mCoutp[4], 256, 1, 0);
  pool(s4h, pingh, 256, 32, 32);
  convL(pingh, mpk[5], mbp[5], s5h,   nullptr, 16, 16, mCinp[5], mCoutp[5], 512, 1, 0);
  pool(s5h, pingh, 512, 16, 16);
  convL(pingh, mpk[6], mbp[6], pongh, nullptr, 8, 8, mCinp[6], mCoutp[6], 512, 1, 0);  // direct path
  upadd(pongh, s5h, pingh, 512, 8, 8);
  convL(pingh, mpk[7], mbp[7], pongh, nullptr, 16, 16, mCinp[7], mCoutp[7], 256, 1, 0);
  upadd(pongh, s4h, pingh, 256, 16, 16);
  convL(pingh, mpk[8], mbp[8], pongh, nullptr, 32, 32, mCinp[8], mCoutp[8], 128, 1, 0);
  upadd(pongh, s3h, pingh, 128, 32, 32);
  convL(pingh, mpk[9], mbp[9], pongh, nullptr, 64, 64, mCinp[9], mCoutp[9], 64, 1, 0);
  upadd(pongh, s2h, pingh, 64, 64, 64);
  convL(pingh, mpk[10], mbp[10], pongh, nullptr, 128, 128, mCinp[10], mCoutp[10], 32, 1, 0);
  upadd(pongh, s1h, pingh, 32, 128, 128);
  convL(pingh, mpk[11], mbp[11], feath, nullptr, 256, 256, mCinp[11], mCoutp[11], 16, 1, 0);

  // ---- branches ----
  convL(feath, bpk[0], bbp[0], pingh, nullptr, 256, 256, bCinp[0], bCoutp[0], 16, 1, 0);
  convL(pingh, bpk[1], bbp[1], nullptr, filt0, 256, 256, bCinp[1], bCoutp[1], 16, 0, 1);
  convL(feath, bpk[2], bbp[2], pingh, nullptr, 256, 256, bCinp[2], bCoutp[2], 16, 1, 0);
  convL(pingh, bpk[3], bbp[3], nullptr, filt1, 256, 256, bCinp[3], bCoutp[3], 16, 0, 1);

  // ---- flow projection ----
  long nBHW = (long)B * HW;
  fill_k<<<eb(nBHW), TPB, 0, stream>>>(sx,  nBHW, 0.f);
  fill_k<<<eb(nBHW), TPB, 0, stream>>>(sy,  nBHW, 0.f);
  fill_k<<<eb(nBHW), TPB, 0, stream>>>(cnt, nBHW, 0.f);
  flow_scatter_k<<<eb(nBHW), TPB, 0, stream>>>(flow_fwd, 0.5f, sx, sy, cnt, B, H, W);
  flow_norm_k<<<eb(nBHW), TPB, 0, stream>>>(sx, sy, cnt, off0, B, HW);
  fill_k<<<eb(nBHW), TPB, 0, stream>>>(sx,  nBHW, 0.f);
  fill_k<<<eb(nBHW), TPB, 0, stream>>>(sy,  nBHW, 0.f);
  fill_k<<<eb(nBHW), TPB, 0, stream>>>(cnt, nBHW, 0.f);
  flow_scatter_k<<<eb(nBHW), TPB, 0, stream>>>(flow_bwd, 0.5f, sx, sy, cnt, B, H, W);
  flow_norm_k<<<eb(nBHW), TPB, 0, stream>>>(sx, sy, cnt, off1, B, HW);

  // ---- adaptive warping + blend ----
  const float* f0 = frames;
  const float* f2 = frames + (long)B * 3 * HW;
  filt_interp_k<<<eb(nBHW), TPB, 0, stream>>>(f0, off0, filt0, w0b, B, H, W);
  filt_interp_k<<<eb(nBHW), TPB, 0, stream>>>(f2, off1, filt1, w2b, B, H, W);

  float* out_cur  = (float*)d_out;
  float* out_rect = (float*)d_out + (long)B * 3 * HW;
  long n3 = (long)B * 3 * HW;
  avg2_k<<<eb(n3), TPB, 0, stream>>>(w0b, w2b, out_cur, n3);

  // ---- rectify input (NHWC f16, Cp=64, channels 41..63 zero) ----
  fill_k<<<eb((long)B * HW * 32), TPB, 0, stream>>>((float*)rinh, (long)B * HW * 32, 0.f);
  copy_ch_nhwc_k<<<eb((long)B * 3  * HW), TPB, 0, stream>>>(out_cur, 0, 3,  rinh, 0,  64, 3,  B, HW);
  copy_ch_nhwc_k<<<eb((long)B * 2  * HW), TPB, 0, stream>>>(off0,    0, 2,  rinh, 3,  64, 2,  B, HW);
  copy_ch_nhwc_k<<<eb((long)B * 2  * HW), TPB, 0, stream>>>(off1,    0, 2,  rinh, 5,  64, 2,  B, HW);
  copy_ch_nhwc_k<<<eb((long)B * 16 * HW), TPB, 0, stream>>>(filt0,   0, 16, rinh, 7,  64, 16, B, HW);
  copy_ch_nhwc_k<<<eb((long)B * 16 * HW), TPB, 0, stream>>>(filt1,   0, 16, rinh, 23, 64, 16, B, HW);
  copy_ch_nhwc_k<<<eb((long)B * 1  * HW), TPB, 0, stream>>>(filt0,   0, 16, rinh, 39, 64, 1,  B, HW);
  copy_ch_nhwc_k<<<eb((long)B * 1  * HW), TPB, 0, stream>>>(filt0,   0, 16, rinh, 40, 64, 1,  B, HW);

  // ---- rectify net ----
  convL(rinh, rpk[0], rbp[0], pingh, nullptr, 256, 256, rCinp[0], rCoutp[0], 64, 1, 0);
  _Float16* ci_ = pingh; _Float16* co_ = pongh;
  for (int i = 1; i < 7; ++i) {
    convL(ci_, rpk[i], rbp[i], co_, nullptr, 256, 256, rCinp[i], rCoutp[i], 64, 1, 0);
    _Float16* tmp = ci_; ci_ = co_; co_ = tmp;
  }
  convL(ci_, rpk[7], rbp[7], nullptr, r3, 256, 256, rCinp[7], rCoutp[7], 3, 0, 1);
  add_k<<<eb(n3), TPB, 0, stream>>>(out_cur, r3, out_rect, n3);

  (void)in_sizes; (void)ws_size; (void)out_size;
}
